// TransformerBlock_22806276342014
// MI455X (gfx1250) — matrix operand order repaired and checked by reading, not compiled, not run on hardware
//
#include <hip/hip_runtime.h>
#include <hip/hip_bf16.h>
#include <math.h>

// ---------------- problem constants ----------------
constexpr int NB  = 8;      // batch
constexpr int NC  = 256;    // channels
constexpr int NC8 = 32;     // qk channels
constexpr int NH  = 128;    // height
constexpr int NW  = 128;    // width
constexpr int NHW = NH * NW;
constexpr int NF  = 1024;   // ffn hidden

typedef __attribute__((ext_vector_type(16))) __bf16 v16bf;
typedef __attribute__((ext_vector_type(8)))  float  v8f;

__device__ inline v8f vzero() { v8f z = {}; return z; }

__device__ inline v8f wmma_bf16(v16bf a, v16bf b, v8f c) {
    // D = A(16x32 bf16) * B(32x16 bf16) + C(16x16 f32)
    return __builtin_amdgcn_wmma_f32_16x16x32_bf16(false, a, false, b, (short)0, c, false, false);
}

// A fragment (16x32 bf16) from row-major bf16 Src[m][k], leading dim ld.
// Per-lane data: 8 contiguous at k0+half*8, 8 contiguous at k0+16+half*8 -> 2x16B loads.
__device__ inline v16bf load_frag_a_bf(const __bf16* __restrict__ src, int ld, int m0, int k0,
                                       int lm, int half) {
    const __bf16* row = src + (size_t)(m0 + lm) * ld + k0 + half * 8;
    union { v16bf v; uint4 q[2]; } u;
    u.q[0] = *(const uint4*)(row);
    u.q[1] = *(const uint4*)(row + 16);
    return u.v;
}

// B fragment (32x16 bf16) from N-major bf16 storage Sn[n][k] (B[k][n] = Sn[n][k]).
// Per-lane data: 8 contiguous at k0+half*8, 8 contiguous at k0+16+half*8 -> 2x16B loads.
__device__ inline v16bf load_frag_b_bf(const __bf16* __restrict__ src, int ld, int n0, int k0,
                                       int lm, int half) {
    const __bf16* row = src + (size_t)(n0 + lm) * ld + k0 + half * 8;
    union { v16bf v; uint4 q[2]; } u;
    u.q[0] = *(const uint4*)(row);
    u.q[1] = *(const uint4*)(row + 16);
    return u.v;
}

// ---- CDNA5 async Global->LDS copy (tracked by ASYNCcnt) ----
// GV mode: per-lane 16B transfer; VDST = LDS byte offset, VADDR = 64-bit addr.
__device__ inline void async_copy_b128(unsigned int lds_off, const void* gaddr) {
    asm volatile("global_load_async_to_lds_b128 %0, %1, off"
                 :
                 : "v"(lds_off), "v"(gaddr)
                 : "memory");
}
__device__ inline void wait_asynccnt0() {
    asm volatile("s_wait_asynccnt 0x0" ::: "memory");
}
__device__ inline unsigned int lds_offset_of(const void* p) {
    // generic LDS pointer: low 32 bits are the LDS byte address (ISA 10.2 aperture rules)
    return (unsigned int)(unsigned long long)p;
}

// =====================================================================
// Prep kernels: one-time weight conversion into bf16 scratch.
// =====================================================================
__global__ void __launch_bounds__(256) cast_bf16_kernel(const float* __restrict__ in,
                                                        __bf16* __restrict__ out, int n) {
    int i = blockIdx.x * 256 + threadIdx.x;
    if (i < n) out[i] = (__bf16)in[i];
}

// in: [R][Cc] row-major f32 ; out: [Cc][R] row-major bf16 (transposed)
__global__ void __launch_bounds__(256) transpose_bf16_kernel(const float* __restrict__ in,
                                                             __bf16* __restrict__ out,
                                                             int R, int Cc) {
    int i = blockIdx.x * 256 + threadIdx.x;
    if (i < R * Cc) {
        int r = i / Cc, c = i % Cc;
        out[(size_t)c * R + r] = (__bf16)in[i];
    }
}

// =====================================================================
// Kernel 1: fused per-(b,h) axial attention.
//   Qt = Xt*Wq^T (+bq+rel_h) ; Kt = Xt*Wk^T (+bk+rel_w)   [stored [w][o]]
//   S  = Qt * Kt^T(B)  -> softmax -> Pb (bf16, row-major [w][v])
//   Tt = Pb * Xs^T(B)  [w][c] ;  AO = Wv * T + bv ; sbuf = x + AO
// LDS: Xs bf16[256][128] | Xst bf16[128][256] | Qs bf16[128][32]
//      | Ks bf16[128][32] | Pm f32[128][128] | Pb bf16[128][128]
//      | Ts bf16[128][256]   = 304 KB
// =====================================================================
__global__ void __launch_bounds__(256) attn_fused_kernel(
    const float* __restrict__ x,
    const __bf16* __restrict__ Wqb, const float* __restrict__ bq,
    const __bf16* __restrict__ Wkb, const float* __restrict__ bk,
    const __bf16* __restrict__ Wvb, const float* __restrict__ bv,
    const float* __restrict__ rel_h, const float* __restrict__ rel_w,
    float* __restrict__ sbuf) {
    extern __shared__ char smem[];
    __bf16* Xs  = (__bf16*)(smem);            // [c=256][w=128]
    __bf16* Xst = (__bf16*)(smem + 65536);    // [w=128][c=256]
    __bf16* Qs  = (__bf16*)(smem + 131072);   // [w=128][o=32]
    __bf16* Ks  = (__bf16*)(smem + 139264);   // [w=128][o=32]
    float*  Pm  = (float*) (smem + 147456);   // [w=128][v=128]
    __bf16* Pb  = (__bf16*)(smem + 212992);   // [w=128][v=128]
    __bf16* Ts  = (__bf16*)(smem + 245760);   // [w=128][c=256]

    const int tid  = threadIdx.x;
    const int lane = tid & 31, lm = lane & 15, half = lane >> 4;
    const int wv   = tid >> 5;                // wave id 0..7
    const int bh   = blockIdx.x;              // 0..B*H-1
    const int b = bh >> 7, h = bh & (NH - 1);

    // ---- Phase 0: stage X slice, both majornesses ----
    {
        const float* xrow = x + (((size_t)(b * NC + tid) * NH + h) * NW);
        __bf16* dst = Xs + (size_t)tid * NW;
#pragma unroll
        for (int w4 = 0; w4 < NW; w4 += 4) {
            float4 v = *(const float4*)(xrow + w4);
            __bf16 b0 = (__bf16)v.x, b1 = (__bf16)v.y, b2 = (__bf16)v.z, b3 = (__bf16)v.w;
            dst[w4 + 0] = b0; dst[w4 + 1] = b1; dst[w4 + 2] = b2; dst[w4 + 3] = b3;
            Xst[(size_t)(w4 + 0) * NC + tid] = b0;
            Xst[(size_t)(w4 + 1) * NC + tid] = b1;
            Xst[(size_t)(w4 + 2) * NC + tid] = b2;
            Xst[(size_t)(w4 + 3) * NC + tid] = b3;
        }
    }
    __syncthreads();

    // ---- Phase 1: Qt/Kt = Xst * W^T  (M=128 w, N=32 o, K=256 c) ----
    {
#pragma unroll
        for (int tt = 0; tt < 4; ++tt) {
            const int t    = wv * 4 + tt;
            const bool isK = (t >= 16);
            const int rem  = t & 15;
            const int mt = rem >> 1, nt = rem & 1;
            const __bf16* Wm = isK ? Wkb : Wqb;
            v8f acc = vzero();
            for (int ks = 0; ks < NC; ks += 32) {
                v16bf a  = load_frag_a_bf(Xst, NC, mt * 16, ks, lm, half);
                v16bf bb = load_frag_b_bf(Wm, NC, nt * 16, ks, lm, half);
                acc = wmma_bf16(a, bb, acc);
            }
#pragma unroll
            for (int r = 0; r < 8; ++r) {
                const int wrow = mt * 16 + half * 8 + r;
                const int o    = nt * 16 + lm;
                if (!isK) {
                    float v = acc[r] + bq[o] + rel_h[o * NH + h];
                    Qs[(size_t)wrow * NC8 + o] = (__bf16)v;
                } else {
                    float v = acc[r] + bk[o] + rel_w[o * NW + wrow];
                    Ks[(size_t)wrow * NC8 + o] = (__bf16)v;
                }
            }
        }
    }
    __syncthreads();

    // ---- Phase 2: S = Qt * Kt^T  (M=128, N=128, K=32: single step) ----
    {
        v16bf a = load_frag_a_bf(Qs, NC8, wv * 16, 0, lm, half);
#pragma unroll
        for (int nt = 0; nt < 8; ++nt) {
            v16bf bb = load_frag_b_bf(Ks, NC8, nt * 16, 0, lm, half);
            v8f acc = vzero();
            acc = wmma_bf16(a, bb, acc);
#pragma unroll
            for (int r = 0; r < 8; ++r)
                Pm[(size_t)(wv * 16 + half * 8 + r) * NW + nt * 16 + lm] = acc[r];
        }
    }
    __syncthreads();

    // ---- Phase 3: row softmax; emit bf16 P once ----
    if (tid < NW) {
        float* row = Pm + (size_t)tid * NW;
        __bf16* rowb = Pb + (size_t)tid * NW;
        float mx = -1e30f;
        for (int j = 0; j < NW; ++j) mx = fmaxf(mx, row[j]);
        float s = 0.f;
        for (int j = 0; j < NW; ++j) { float e = expf(row[j] - mx); row[j] = e; s += e; }
        float inv = 1.f / s;
        for (int j = 0; j < NW; ++j) rowb[j] = (__bf16)(row[j] * inv);
    }
    __syncthreads();

    // ---- Phase 4: Tt = P * X^T  (M=128 w, N=256 c, K=128 v) ----
    {
        v8f acc[16];
#pragma unroll
        for (int nt = 0; nt < 16; ++nt) acc[nt] = vzero();
        for (int ks = 0; ks < NW; ks += 32) {
            v16bf a = load_frag_a_bf(Pb, NW, wv * 16, ks, lm, half);
#pragma unroll
            for (int nt = 0; nt < 16; ++nt) {
                v16bf bb = load_frag_b_bf(Xs, NW, nt * 16, ks, lm, half);
                acc[nt] = wmma_bf16(a, bb, acc[nt]);
            }
        }
#pragma unroll
        for (int nt = 0; nt < 16; ++nt)
#pragma unroll
            for (int r = 0; r < 8; ++r) {
                const int wrow = wv * 16 + half * 8 + r;
                Ts[(size_t)wrow * NC + nt * 16 + lm] = (__bf16)acc[nt][r];
            }
    }
    __syncthreads();

    // ---- Phase 5: AO = Wv * T + bv ; sbuf = x + AO ----
    {
        v8f acc[2][8];
#pragma unroll
        for (int mi = 0; mi < 2; ++mi)
#pragma unroll
            for (int nt = 0; nt < 8; ++nt) acc[mi][nt] = vzero();
        for (int ks = 0; ks < NC; ks += 32) {
#pragma unroll
            for (int mi = 0; mi < 2; ++mi) {
                v16bf a = load_frag_a_bf(Wvb, NC, (wv * 2 + mi) * 16, ks, lm, half);
#pragma unroll
                for (int nt = 0; nt < 8; ++nt) {
                    v16bf bb = load_frag_b_bf(Ts, NC, nt * 16, ks, lm, half);
                    acc[mi][nt] = wmma_bf16(a, bb, acc[mi][nt]);
                }
            }
        }
#pragma unroll
        for (int mi = 0; mi < 2; ++mi)
#pragma unroll
            for (int nt = 0; nt < 8; ++nt)
#pragma unroll
                for (int r = 0; r < 8; ++r) {
                    const int c  = (wv * 2 + mi) * 16 + half * 8 + r;
                    const int wc = nt * 16 + lm;
                    const size_t gi = ((size_t)(b * NC + c) * NH + h) * NW + wc;
                    sbuf[gi] = acc[mi][nt][r] + bv[c] + x[gi];
                }
    }
}

// =====================================================================
// Kernel 2: LN1 over 256-contiguous-element chunks of sbuf (in place).
// =====================================================================
__global__ void __launch_bounds__(256) ln1_chunk_kernel(float* __restrict__ sb,
                                                        const float* __restrict__ g,
                                                        const float* __restrict__ be) {
    const int tid = threadIdx.x;
    const size_t base = (size_t)blockIdx.x * 256;
    float v = sb[base + tid];
    float s = v, q = v * v;
#pragma unroll
    for (int off = 16; off > 0; off >>= 1) {
        s += __shfl_down(s, off);
        q += __shfl_down(q, off);
    }
    __shared__ float rs[8], rq[8];
    __shared__ float mean_s, inv_s;
    if ((tid & 31) == 0) { rs[tid >> 5] = s; rq[tid >> 5] = q; }
    __syncthreads();
    if (tid == 0) {
        float S = 0.f, Q = 0.f;
#pragma unroll
        for (int i = 0; i < 8; ++i) { S += rs[i]; Q += rq[i]; }
        float m   = S * (1.f / 256.f);
        float var = Q * (1.f / 256.f) - m * m;
        mean_s = m;
        inv_s  = rsqrtf(var + 1e-5f);
    }
    __syncthreads();
    sb[base + tid] = (v - mean_s) * inv_s * g[tid] + be[tid];
}

// =====================================================================
// Kernel 3: FFN + LN2, 128-row tile per block of x3[r,c] = x2[b,c,p],
// r = p*8 + b.  Weight staging uses async Global->LDS (ASYNCcnt).
// LDS: Xt bf16[128][256] | W1t bf16[128][256] | Hs bf16[128][128]
//      | W2t bf16[256][128]  = 224 KB
// =====================================================================
__global__ void __launch_bounds__(256) ffn_ln2_kernel(
    const float* __restrict__ x2,             // sbuf viewed [B][C][HW]
    const __bf16* __restrict__ W1t_g,         // [F][C] bf16
    const float* __restrict__ b1,
    const __bf16* __restrict__ W2t_g,         // [C][F] bf16
    const float* __restrict__ b2,
    const float* __restrict__ g2, const float* __restrict__ be2,
    float* __restrict__ out) {
    extern __shared__ char smem[];
    __bf16* Xt  = (__bf16*)(smem);            // [r=128][c=256]
    __bf16* W1t = (__bf16*)(smem + 65536);    // [f=128][c=256]
    __bf16* Hs  = (__bf16*)(smem + 131072);   // [r=128][f=128]
    __bf16* W2t = (__bf16*)(smem + 163840);   // [c=256][f=128]

    const int tid  = threadIdx.x;
    const int lane = tid & 31, lm = lane & 15, half = lane >> 4;
    const int wv   = tid >> 5;
    const int blk  = blockIdx.x;              // 1024 blocks of 128 rows
    const size_t p0 = (size_t)blk * 16;

    const unsigned int w1_lds = lds_offset_of(W1t);
    const unsigned int w2_lds = lds_offset_of(W2t);

    // ---- gather x3 tile: Xt[pl*8+bb][c] = x2[(bb*C+c)*HW + p0+pl] ----
    for (int idx = tid; idx < 2048; idx += 256) {
        const int c = idx >> 3, bb = idx & 7;
        const float* src = x2 + ((size_t)(bb * NC + c) * NHW) + p0;
#pragma unroll
        for (int pl = 0; pl < 16; pl += 4) {
            float4 v = *(const float4*)(src + pl);
            Xt[(size_t)((pl + 0) * 8 + bb) * NC + c] = (__bf16)v.x;
            Xt[(size_t)((pl + 1) * 8 + bb) * NC + c] = (__bf16)v.y;
            Xt[(size_t)((pl + 2) * 8 + bb) * NC + c] = (__bf16)v.z;
            Xt[(size_t)((pl + 3) * 8 + bb) * NC + c] = (__bf16)v.w;
        }
    }

    v8f acc2[16];
#pragma unroll
    for (int nt = 0; nt < 16; ++nt) acc2[nt] = vzero();

    for (int fc = 0; fc < NF; fc += 128) {
        __syncthreads();   // previous chunk fully consumed before restaging
        // stage W1^T chunk (contiguous 64 KB) + W2^T chunk (256 rows x 256 B)
        // via async Global->LDS b128 transfers (no VGPR round-trip).
        {
            const uint4* s1 = (const uint4*)(W1t_g + (size_t)fc * NC);
#pragma unroll
            for (int i = 0; i < 16; ++i) {
                const int e = tid + i * 256;                  // 4096 x 16B
                async_copy_b128(w1_lds + e * 16, s1 + e);
            }
#pragma unroll
            for (int i = 0; i < 16; ++i) {
                const int e = tid + i * 256;                  // 4096 x 16B
                const int c = e >> 4, j = e & 15;
                async_copy_b128(w2_lds + (unsigned)(c * 256 + j * 16),
                                (const uint4*)(W2t_g + (size_t)c * NF + fc) + j);
            }
            wait_asynccnt0();
        }
        // prefetch next chunk while we compute on this one
        if (fc + 128 < NF) {
            __builtin_prefetch((const char*)(W1t_g + (size_t)(fc + 128) * NC) + tid * 256, 0, 1);
            __builtin_prefetch((const char*)(W2t_g + (size_t)tid * NF + fc + 128), 0, 1);
        }
        __syncthreads();

        // GEMM1: wave wv owns rows 16wv..16wv+15, N=128 chunk, K=256
        v8f acc1[8];
#pragma unroll
        for (int nt = 0; nt < 8; ++nt) acc1[nt] = vzero();
        for (int ks = 0; ks < NC; ks += 32) {
            v16bf a = load_frag_a_bf(Xt, NC, wv * 16, ks, lm, half);
#pragma unroll
            for (int nt = 0; nt < 8; ++nt) {
                v16bf bb = load_frag_b_bf(W1t, NC, nt * 16, ks, lm, half);
                acc1[nt] = wmma_bf16(a, bb, acc1[nt]);
            }
        }
        // exact GELU -> Hs
#pragma unroll
        for (int nt = 0; nt < 8; ++nt)
#pragma unroll
            for (int r = 0; r < 8; ++r) {
                const int m  = wv * 16 + half * 8 + r;
                const int fl = nt * 16 + lm;
                float hv = acc1[nt][r] + b1[fc + fl];
                float gl = 0.5f * hv * (1.f + erff(hv * 0.70710678118654752f));
                Hs[(size_t)m * 128 + fl] = (__bf16)gl;
            }
        __syncthreads();

        // GEMM2 accumulate: A=Hs[m][f], B=W2t (N-major), K=128 chunk
        for (int ks = 0; ks < 128; ks += 32) {
            v16bf a = load_frag_a_bf(Hs, 128, wv * 16, ks, lm, half);
#pragma unroll
            for (int nt = 0; nt < 16; ++nt) {
                v16bf bb = load_frag_b_bf(W2t, 128, nt * 16, ks, lm, half);
                acc2[nt] = wmma_bf16(a, bb, acc2[nt]);
            }
        }
    }

    // ---- epilogue: +b2, +residual (re-gathered), LN2, store ----
#pragma unroll
    for (int r = 0; r < 8; ++r) {
        const int m = wv * 16 + half * 8 + r;
        const size_t rg = (size_t)blk * 128 + m;
        const int bb = (int)(rg & 7);
        const size_t p = rg >> 3;
        float vals[16];
        float s = 0.f, q = 0.f;
#pragma unroll
        for (int nt = 0; nt < 16; ++nt) {
            const int c = nt * 16 + lm;
            float t = acc2[nt][r] + b2[c] + x2[(size_t)(bb * NC + c) * NHW + p];
            vals[nt] = t;
            s += t;
            q += t * t;
        }
#pragma unroll
        for (int off = 1; off < 16; off <<= 1) {   // stays within 16-lane half
            s += __shfl_xor(s, off);
            q += __shfl_xor(q, off);
        }
        const float mean = s * (1.f / 256.f);
        const float var  = q * (1.f / 256.f) - mean * mean;
        const float inv  = rsqrtf(var + 1e-5f);
#pragma unroll
        for (int nt = 0; nt < 16; ++nt) {
            const int c = nt * 16 + lm;
            out[rg * NC + c] = (vals[nt] - mean) * inv * g2[c] + be2[c];
        }
    }
}

// =====================================================================
extern "C" void kernel_launch(void* const* d_in, const int* in_sizes, int n_in,
                              void* d_out, int out_size, void* d_ws, size_t ws_size,
                              hipStream_t stream) {
    (void)in_sizes; (void)n_in; (void)out_size; (void)ws_size;
    const float* x     = (const float*)d_in[0];
    const float* Wq    = (const float*)d_in[1];
    const float* bq    = (const float*)d_in[2];
    const float* Wk    = (const float*)d_in[3];
    const float* bk    = (const float*)d_in[4];
    const float* Wv    = (const float*)d_in[5];
    const float* bv    = (const float*)d_in[6];
    const float* rel_h = (const float*)d_in[7];
    const float* rel_w = (const float*)d_in[8];
    const float* W1    = (const float*)d_in[9];
    const float* b1    = (const float*)d_in[10];
    const float* W2    = (const float*)d_in[11];
    const float* b2    = (const float*)d_in[12];
    const float* g1    = (const float*)d_in[13];
    const float* be1   = (const float*)d_in[14];
    const float* g2    = (const float*)d_in[15];
    const float* be2   = (const float*)d_in[16];

    // workspace layout
    float* sbuf = (float*)d_ws;                       // [B][C][H][W] f32 (134 MB)
    char*  wb   = (char*)d_ws + 134217728;
    __bf16* Wqb  = (__bf16*)(wb);                     // [32][256]
    __bf16* Wkb  = (__bf16*)(wb + 16384);             // [32][256]
    __bf16* Wvb  = (__bf16*)(wb + 32768);             // [256][256]
    __bf16* W1tg = (__bf16*)(wb + 163840);            // [1024][256] = W1^T
    __bf16* W2tg = (__bf16*)(wb + 688128);            // [256][1024] = W2^T

    // ---- weight prep (bf16 cast + transposes) ----
    cast_bf16_kernel<<<(NC8 * NC + 255) / 256, 256, 0, stream>>>(Wq, Wqb, NC8 * NC);
    cast_bf16_kernel<<<(NC8 * NC + 255) / 256, 256, 0, stream>>>(Wk, Wkb, NC8 * NC);
    cast_bf16_kernel<<<(NC * NC + 255) / 256, 256, 0, stream>>>(Wv, Wvb, NC * NC);
    transpose_bf16_kernel<<<(NC * NF + 255) / 256, 256, 0, stream>>>(W1, W1tg, NC, NF);
    transpose_bf16_kernel<<<(NF * NC + 255) / 256, 256, 0, stream>>>(W2, W2tg, NF, NC);

    const size_t attn_smem = 311296;  // 304 KB
    const size_t ffn_smem  = 229376;  // 224 KB

    attn_fused_kernel<<<NB * NH, 256, attn_smem, stream>>>(
        x, Wqb, bq, Wkb, bk, Wvb, bv, rel_h, rel_w, sbuf);

    ln1_chunk_kernel<<<(NB * NC * NH * NW) / 256, 256, 0, stream>>>(sbuf, g1, be1);

    ffn_ln2_kernel<<<(NHW * NB) / 128, 256, ffn_smem, stream>>>(
        sbuf, W1tg, b1, W2tg, b2, g2, be2, (float*)d_out);
}
